// VectorQuantizer_73718818668649
// MI455X (gfx1250) — matrix operand order repaired and checked by reading, not compiled, not run on hardware
//
#include <hip/hip_runtime.h>
#include <hip/hip_bf16.h>
#include <math.h>

typedef __attribute__((ext_vector_type(16))) _Float16 v16h;
typedef __attribute__((ext_vector_type(8)))  _Float16 v8h;
typedef __attribute__((ext_vector_type(8)))  float    v8f;
typedef __attribute__((ext_vector_type(2)))  float    v2f;

#define B_     32
#define C_     64
#define L_     512
#define VOCAB_ 8192
#define NEL_   (B_ * C_ * L_)   // 1048576

// ---------------------------------------------------------------------------
// Normalize codebook rows and store as f16 in WMMA B-fragment layout.
// B (32x16 f16): lane n (0..15) holds column N=n, halfs 0..7 = K 0..7, 8..15 = K 16..23;
// lane n+16 holds column N=n, halfs 0..7 = K 8..15, 8..15 = K 24..31. Two fragments per
// 16-vocab chunk cover K=0..31 and K=32..63. Chunk vc is 1024 contiguous
// halfs at cbf + vc*1024 (frag0 then frag1).
// ---------------------------------------------------------------------------
__global__ void prep_cb_kernel(const float* __restrict__ embed,
                               _Float16* __restrict__ cbf) {
    int v = blockIdx.x * blockDim.x + threadIdx.x;
    if (v >= VOCAB_) return;
    const float* row = embed + v * C_;
    float ss = 0.f;
#pragma unroll 8
    for (int k = 0; k < C_; ++k) { float x = row[k]; ss += x * x; }
    float nrm = fmaxf(sqrtf(ss), 1e-12f);
    float inv = 1.0f / nrm;
    int vc = v >> 4, n = v & 15;
    for (int k = 0; k < C_; ++k) {
        int frag = k >> 5, kk = k & 31;
        int lane = n + (((kk >> 3) & 1) << 4);
        int h    = (kk & 7) + ((kk >> 4) << 3);
        cbf[(((vc * 2 + frag) * 32 + lane) << 4) + h] = (_Float16)(row[k] * inv);
    }
}

// ---------------------------------------------------------------------------
// f_rest = f ; f_hat (= d_out) = 0 ; loss accumulators = 0
// ---------------------------------------------------------------------------
__global__ void init_kernel(const float* __restrict__ f,
                            float* __restrict__ f_rest,
                            float* __restrict__ f_hat,
                            float* __restrict__ lossAcc) {
    int i = blockIdx.x * blockDim.x + threadIdx.x;
    if (i < NEL_) { float x = f[i]; f_rest[i] = x; f_hat[i] = 0.f; }
    if (i < 16) lossAcc[i] = 0.f;
    if (i == 0) f_hat[NEL_] = 0.f;   // loss slot of d_out
}

// ---------------------------------------------------------------------------
// Block-mean pool of f_rest to [B, s, C] tokens, converted to f16 (t-major).
// ---------------------------------------------------------------------------
__global__ void pool_kernel(const float* __restrict__ f_rest,
                            _Float16* __restrict__ X, int s, int blk) {
    int n = blockIdx.x * blockDim.x + threadIdx.x;
    int total = B_ * s * C_;
    if (n >= total) return;
    int t = n >> 6, c = n & 63;
    int b = t / s, j = t - b * s;
    const float* src = f_rest + (((b * C_) + c) << 9) + j * blk;
    float sum = 0.f;
    for (int i = 0; i < blk; ++i) sum += src[i];
    X[(t << 6) + c] = (_Float16)(sum * (1.0f / (float)blk));
}

// ---------------------------------------------------------------------------
// Cosine-sim argmax via WMMA with LDS-blocked codebook streaming.
// Block = 128 threads = 4 waves; each wave owns MT=2 token tiles (A frags in
// registers). All 4 waves share the codebook stream: 8 vocab chunks (16 KB)
// are staged to LDS per outer iteration, then each wave reads B fragments
// with ds_load_b128 and issues 2 WMMAs per tile per chunk (K=64 = 2x32).
// C/D layout: lanes 0-15 -> (M=j, N=lane), lanes 16-31 -> (M=j+8, N=lane-16).
// ---------------------------------------------------------------------------
#define MT 2     // token tiles per wave
#define OC 8     // vocab chunks staged per outer iteration (16 KB)

__global__ void __launch_bounds__(128)
argmax_kernel(const _Float16* __restrict__ X,
              const _Float16* __restrict__ cbf,
              int* __restrict__ idxout, int ntiles) {
    __shared__ __align__(16) _Float16 cbs[OC * 1024];

    int tid  = threadIdx.x;
    int lane = tid & 31;
    int wv   = tid >> 5;
    int t0   = (blockIdx.x * 4 + wv) * MT;

    int m  = lane & 15;
    int hi = lane >> 4;

    // A fragments (16-bit A 16x32): lane<16: halfs = K0..7,K16..23;
    // lane>=16: halfs = K8..15,K24..31. Second fragment: +32.
    v16h A0[MT], A1[MT];
#pragma unroll
    for (int q = 0; q < MT; ++q) {
        int t = t0 + q; if (t >= ntiles) t = 0;   // clamp; results discarded
        const _Float16* row = X + ((t * 16 + m) << 6);
        v8h a0lo = *(const v8h*)(row + hi * 8);
        v8h a0hi = *(const v8h*)(row + 16 + hi * 8);
        v8h a1lo = *(const v8h*)(row + 32 + hi * 8);
        v8h a1hi = *(const v8h*)(row + 48 + hi * 8);
        A0[q] = __builtin_shufflevector(a0lo, a0hi,
                    0,1,2,3,4,5,6,7,8,9,10,11,12,13,14,15);
        A1[q] = __builtin_shufflevector(a1lo, a1hi,
                    0,1,2,3,4,5,6,7,8,9,10,11,12,13,14,15);
    }

    float best[MT][8]; int bidx[MT][8];
#pragma unroll
    for (int q = 0; q < MT; ++q)
#pragma unroll
        for (int j = 0; j < 8; ++j) { best[q][j] = -3.4e38f; bidx[q][j] = 0; }

    const uint4* src = (const uint4*)cbf;   // 16B units; one chunk = 128 uint4
    uint4*       dst = (uint4*)cbs;

    for (int outer = 0; outer < (VOCAB_ / 16) / OC; ++outer) {
        int base = outer * (OC * 128);
        // cooperative staging: 16 KB contiguous
#pragma unroll
        for (int k = 0; k < OC; ++k)
            dst[tid + k * 128] = src[base + tid + k * 128];
        if (outer + 1 < (VOCAB_ / 16) / OC)   // prefetch next outer block
            __builtin_prefetch((const char*)(src + base + OC * 128) + tid * 128, 0, 3);
        __syncthreads();

#pragma unroll 2
        for (int cc = 0; cc < OC; ++cc) {
            const _Float16* cp = cbs + cc * 1024;
            v16h b0 = *(const v16h*)(cp + (lane << 4));
            v16h b1 = *(const v16h*)(cp + ((32 + lane) << 4));
            int nidx = (outer * OC + cc) * 16 + (lane & 15);
#pragma unroll
            for (int q = 0; q < MT; ++q) {
                v8f c = {};
                c = __builtin_amdgcn_wmma_f32_16x16x32_f16(false, A0[q], false, b0,
                                                           (short)0, c, false, false);
                c = __builtin_amdgcn_wmma_f32_16x16x32_f16(false, A1[q], false, b1,
                                                           (short)0, c, false, false);
#pragma unroll
                for (int j = 0; j < 8; ++j) {
                    float v = c[j];
                    if (v > best[q][j]) {      // strict > keeps first max
                        best[q][j] = v; bidx[q][j] = nidx;
                    }
                }
            }
        }
        __syncthreads();
    }

    // Cross-lane argmax reduce within each 16-lane half; min index on ties.
#pragma unroll
    for (int q = 0; q < MT; ++q) {
#pragma unroll
        for (int off = 8; off >= 1; off >>= 1) {
#pragma unroll
            for (int j = 0; j < 8; ++j) {
                float ov = __shfl_xor(best[q][j], off, 32);
                int   oi = __shfl_xor(bidx[q][j], off, 32);
                if (ov > best[q][j] || (ov == best[q][j] && oi < bidx[q][j])) {
                    best[q][j] = ov; bidx[q][j] = oi;
                }
            }
        }
        if ((lane & 15) == 0 && (t0 + q) < ntiles) {
            int mbase = (t0 + q) * 16 + (hi << 3);
#pragma unroll
            for (int j = 0; j < 8; ++j) idxout[mbase + j] = bidx[q][j];
        }
    }
}

// ---------------------------------------------------------------------------
// Gather h = embed[idx] (repeat-upsampled), phi: 0.5*h + 0.5*(conv1d(h)+b),
// f_hat += , f_rest -= , squared-error accumulation.
// Conv as implicit GEMM on the f32 matrix pipe (v_wmma_f32_16x16x4_f32),
// K padded to 4 per input channel (kk=3 weight zero-staged).
// ---------------------------------------------------------------------------
#define HS_N   (64 * 132)
#define WS4_N  (64 * 64 * 4)

__global__ void __launch_bounds__(256)
fuse_kernel(const float* __restrict__ f, const float* __restrict__ embed,
            const float* __restrict__ w, const float* __restrict__ bias,
            const int* __restrict__ idx, float* __restrict__ f_rest,
            float* __restrict__ f_hat, float* __restrict__ lossAcc,
            int s, int shift, int si) {
    extern __shared__ float smem[];
    float* hs  = smem;               // [64][132], p=0 is l = lbase0-1
    float* ws4 = hs + HS_N;          // [(ci*64+co)*4 + kk], kk=3 -> 0
    float* bs  = ws4 + WS4_N;        // [64]
    float* red = bs + 64;            // [256]

    int tid    = threadIdx.x;
    int b      = blockIdx.x >> 2;
    int lbase0 = (blockIdx.x & 3) * 128;

    for (int i = tid; i < 64 * 130; i += 256) {
        int ci = i / 130, p = i - ci * 130;
        int l  = lbase0 + p - 1;
        float v = 0.f;
        if (l >= 0 && l < L_) v = embed[(idx[b * s + (l >> shift)] << 6) + ci];
        hs[ci * 132 + p] = v;
    }
    for (int i = tid; i < WS4_N; i += 256) {
        int kk = i & 3, co = (i >> 2) & 63, ci = i >> 8;
        ws4[i] = (kk < 3) ? w[(((co << 6) + ci) * 3) + kk] : 0.f;
    }
    if (tid < 64) bs[tid] = bias[tid];
    __syncthreads();

    int wavei = tid >> 5;
    int lane  = tid & 31;
    int n     = lane & 15;
    int hi    = lane >> 4;

    float local = 0.f;
    for (int tt = wavei; tt < 32; tt += 8) {
        int ct = tt & 3;
        int lt = tt >> 2;
        int pc = lt * 16 + n + 1;
        int offB0 = pc - 1 + 2 * hi;  // K0 (lo: l-1) / K2 (hi: l+1)
        int offB1 = pc;               // K1 (lo: l)   / K3 dummy (weight 0)
        int cobase = ct * 16;
        int wlane  = cobase + n;

        v8f c;
#pragma unroll
        for (int j = 0; j < 8; ++j) c[j] = bs[cobase + j + 8 * hi];

#if __has_builtin(__builtin_amdgcn_wmma_f32_16x16x4_f32)
#pragma unroll 4
        for (int ci = 0; ci < 64; ++ci) {
            const float* hrow = hs + ci * 132;
            v2f Bv = { hrow[offB0], hrow[offB1] };
            v2f Av = *(const v2f*)(ws4 + ((((ci << 6) + wlane) << 2) + 2 * hi));
            c = __builtin_amdgcn_wmma_f32_16x16x4_f32(false, Av, false, Bv,
                                                      (short)0, c, false, false);
        }
#else
        for (int ci = 0; ci < 64; ++ci) {
            const float* hrow = hs + ci * 132;
            float h0 = hrow[pc - 1], h1 = hrow[pc], h2 = hrow[pc + 1];
#pragma unroll
            for (int j = 0; j < 8; ++j) {
                const float* wp = ws4 + (((ci << 6) + cobase + j + 8 * hi) << 2);
                c[j] += wp[0] * h0 + wp[1] * h1 + wp[2] * h2;
            }
        }
#endif

        int lout = lbase0 + lt * 16 + n;
#pragma unroll
        for (int j = 0; j < 8; ++j) {
            int co = cobase + j + 8 * hi;
            float hcen = hs[co * 132 + pc];
            float outv = 0.5f * hcen + 0.5f * c[j];   // h*(1-RESI)+(conv+b)*RESI
            int g = (((b << 6) + co) << 9) + lout;
            float fh = f_hat[g] + outv;
            f_hat[g] = fh;
            f_rest[g] -= outv;
            float d = fh - f[g];
            local += d * d;
        }
    }

    red[tid] = local;
    __syncthreads();
    for (int st = 128; st > 0; st >>= 1) {
        if (tid < st) red[tid] += red[tid + st];
        __syncthreads();
    }
    if (tid == 0) atomicAdd(lossAcc + si, red[0]);
}

__global__ void finalize_kernel(const float* __restrict__ lossAcc,
                                float* __restrict__ out) {
    float loss = 0.f;
    for (int si = 0; si < 10; ++si)
        loss += 1.25f * lossAcc[si] * (1.0f / (float)NEL_);
    out[NEL_] = loss * 0.1f;
}

// ---------------------------------------------------------------------------
extern "C" void kernel_launch(void* const* d_in, const int* in_sizes, int n_in,
                              void* d_out, int out_size, void* d_ws, size_t ws_size,
                              hipStream_t stream) {
    const float* f      = (const float*)d_in[0];
    const float* embed  = (const float*)d_in[1];
    const float* phi_w  = (const float*)d_in[2];
    const float* phi_b  = (const float*)d_in[3];
    float* out = (float*)d_out;

    // Workspace layout (< 8 MB total)
    float*    f_rest  = (float*)d_ws;                                  // 4 MB
    _Float16* X       = (_Float16*)((char*)d_ws + (4u << 20));         // 2 MB
    _Float16* cbf     = (_Float16*)((char*)d_ws + (6u << 20));         // 1 MB
    int*      idxbuf  = (int*)((char*)d_ws + (7u << 20));              // 64 KB
    float*    lossAcc = (float*)((char*)d_ws + (7u << 20) + (64u << 10));

    prep_cb_kernel<<<VOCAB_ / 256, 256, 0, stream>>>(embed, cbf);
    init_kernel<<<(NEL_ + 255) / 256, 256, 0, stream>>>(f, f_rest, out, lossAcc);

    const size_t smemBytes = (size_t)(HS_N + WS4_N + 64 + 256) * sizeof(float);

    static const int pidx_tab[10] = {0, 0, 0, 1, 1, 2, 2, 2, 3, 3};
    for (int si = 0; si < 10; ++si) {
        int s     = 1 << si;
        int N     = B_ * s;
        int blk   = L_ / s;
        int shift = 9 - si;

        int totalPool = N * C_;
        pool_kernel<<<(totalPool + 255) / 256, 256, 0, stream>>>(f_rest, X, s, blk);

        int ntiles  = N / 16;
        int nblocks = (ntiles + 4 * MT - 1) / (4 * MT);
        argmax_kernel<<<nblocks, 128, 0, stream>>>(X, cbf, idxbuf, ntiles);

        const float* wp = phi_w + pidx_tab[si] * C_ * C_ * 3;
        const float* bb = phi_b + pidx_tab[si] * C_;
        fuse_kernel<<<B_ * 4, 256, smemBytes, stream>>>(f, embed, wp, bb, idxbuf,
                                                        f_rest, out, lossAcc,
                                                        s, shift, si);
    }
    finalize_kernel<<<1, 1, 0, stream>>>(lossAcc, out);
}
